// SNNClassifier_21818433864395
// MI455X (gfx1250) — matrix operand order repaired and checked by reading, not compiled, not run on hardware
//
#include <hip/hip_runtime.h>

typedef __attribute__((ext_vector_type(16))) __bf16 v16bf;
typedef __attribute__((ext_vector_type(8)))  __bf16 v8bf;
typedef __attribute__((ext_vector_type(8)))  float  v8f;
typedef __attribute__((ext_vector_type(4)))  float  f4;
typedef __attribute__((ext_vector_type(4)))  __bf16 bf4;

#define B_  256
#define T_  128
#define I_  1024
#define H_  2048
#define TT  8        // timesteps processed jointly (B-fragment reuse)

// ---------------------------------------------------------------------------
// Convert x [B,T,I] fp32 -> Xb [T,B,I] bf16 (transpose so per-t A tiles are
// contiguous across the batch dimension), vectorized 4-wide along I.
// ---------------------------------------------------------------------------
__global__ void k_convert_x(const float* __restrict__ x, __bf16* __restrict__ xb) {
    const long TI = (long)T_ * I_;
    const long n4 = (long)B_ * T_ * I_ / 4;
    for (long v = (long)blockIdx.x * blockDim.x + threadIdx.x; v < n4;
         v += (long)gridDim.x * blockDim.x) {
        long n   = v * 4;
        long b   = n / TI;
        long rem = n - b * TI;
        long t   = rem / I_;
        long i   = rem - t * I_;
        f4 f = *(const f4*)(x + n);
        bf4 o;
        o[0] = (__bf16)f[0]; o[1] = (__bf16)f[1];
        o[2] = (__bf16)f[2]; o[3] = (__bf16)f[3];
        *(bf4*)(xb + ((t * B_ + b) * I_ + i)) = o;
    }
}

// w1 [H,I] fp32 -> bf16, same layout
__global__ void k_convert_w(const float* __restrict__ w, __bf16* __restrict__ wb) {
    const long n4 = (long)H_ * I_ / 4;
    for (long v = (long)blockIdx.x * blockDim.x + threadIdx.x; v < n4;
         v += (long)gridDim.x * blockDim.x) {
        long n = v * 4;
        f4 f = *(const f4*)(w + n);
        bf4 o;
        o[0] = (__bf16)f[0]; o[1] = (__bf16)f[1];
        o[2] = (__bf16)f[2]; o[3] = (__bf16)f[3];
        *(bf4*)(wb + n) = o;
    }
}

// ---------------------------------------------------------------------------
// Fused SNN recurrence. Each workgroup: 32(B) x 64(H) membrane tile.
// 8 waves -> each wave one 16x16 WMMA C-tile; mem & spike-count held in the
// native C layout (8 VGPRs each). Time blocked by TT so each w1 B-fragment
// is loaded once and reused across TT independent WMMAs.
// ---------------------------------------------------------------------------
__global__ __launch_bounds__(256) void
k_snn(const __bf16* __restrict__ Xb,   // [T, B, I] bf16
      const __bf16* __restrict__ Wb,   // [H, I]    bf16
      const float*  __restrict__ b1,   // [H]
      float*        __restrict__ S)    // [B, H] spike counts
{
    const int lane = threadIdx.x & 31;
    const int wave = threadIdx.x >> 5;        // 0..7
    const int wm   = wave & 1;                 // M sub-tile (0..1)
    const int wn   = wave >> 1;                // N sub-tile (0..3)
    const int l16  = lane & 15;
    const int lhi  = lane >> 4;                // 0 / 1

    const int bBase = blockIdx.x * 32 + wm * 16;      // batch-row base of tile
    const int nBase = blockIdx.y * 64 + wn * 16;      // H-column base of tile

    // A-fragment (16x32 bf16): lane<16 -> row M=l16, K in {0..7,16..23};
    //                          lane>=16 -> same rows, K in {8..15,24..31}
    const int rowA  = bBase + l16;
    const int kOffA = lhi * 8;
    // B-fragment (32x16 bf16): lane<16 -> col N=l16, K in {0..7,16..23}; lane>=16 K in {8..15,24..31}
    const int colB  = nBase + l16;
    const int kOffB = lhi * 8;

    const float bias = b1[colB];
    const __bf16* wrow = Wb + (size_t)colB * I_ + kOffB;

    v8f mem, cnt;
#pragma unroll
    for (int r = 0; r < 8; ++r) { mem[r] = 0.0f; cnt[r] = 0.0f; }

    for (int t0 = 0; t0 < T_; t0 += TT) {
        v8f c[TT];
#pragma unroll
        for (int tt = 0; tt < TT; ++tt)
#pragma unroll
            for (int r = 0; r < 8; ++r) c[tt][r] = 0.0f;

        for (int k = 0; k < I_; k += 32) {
            // load B fragment once, reuse for TT timesteps
            v8bf blo = *(const v8bf*)(wrow + k);
            v8bf bhi = *(const v8bf*)(wrow + k + 16);
            v16bf bf = __builtin_shufflevector(blo, bhi,
                0,1,2,3,4,5,6,7,8,9,10,11,12,13,14,15);
#pragma unroll
            for (int tt = 0; tt < TT; ++tt) {
                const __bf16* xrow =
                    Xb + ((size_t)(t0 + tt) * B_ + rowA) * I_ + kOffA;
                v8bf alo = *(const v8bf*)(xrow + k);
                v8bf ahi = *(const v8bf*)(xrow + k + 16);
                v16bf af = __builtin_shufflevector(alo, ahi,
                    0,1,2,3,4,5,6,7,8,9,10,11,12,13,14,15);
                c[tt] = __builtin_amdgcn_wmma_f32_16x16x32_bf16(
                            false, af, false, bf, (short)0, c[tt], false, false);
            }
        }

        // sequential leaky-integrate-and-fire over the TT timesteps
#pragma unroll
        for (int tt = 0; tt < TT; ++tt) {
#pragma unroll
            for (int r = 0; r < 8; ++r) {
                float cur   = c[tt][r] + bias;
                float m     = mem[r];
                float reset = (m > 1.0f) ? 1.0f : 0.0f;   // from PREVIOUS mem
                m = 0.95f * m + cur - reset;               // THRESH = 1
                mem[r] = m;
                cnt[r] += (m > 1.0f) ? 1.0f : 0.0f;        // spike
            }
        }
    }

    // C layout: lanes 0-15 -> M=r, lanes 16-31 -> M=r+8; N = l16
    const int row0 = bBase + lhi * 8;
#pragma unroll
    for (int r = 0; r < 8; ++r)
        S[(size_t)(row0 + r) * H_ + colB] = cnt[r];
}

// ---------------------------------------------------------------------------
// out[b,o] = sum_h S[b,h] * w2[o,h] + T * b2[o]   (O = 2)
// ---------------------------------------------------------------------------
__global__ void k_out(const float* __restrict__ S, const float* __restrict__ w2,
                      const float* __restrict__ b2, float* __restrict__ out) {
    __shared__ float red0[256];
    __shared__ float red1[256];
    const int b   = blockIdx.x;
    const int tid = threadIdx.x;
    float p0 = 0.0f, p1 = 0.0f;
    for (int h = tid; h < H_; h += 256) {
        float s = S[(size_t)b * H_ + h];
        p0 += s * w2[h];
        p1 += s * w2[H_ + h];
    }
    red0[tid] = p0; red1[tid] = p1;
    __syncthreads();
    for (int off = 128; off > 0; off >>= 1) {
        if (tid < off) { red0[tid] += red0[tid + off]; red1[tid] += red1[tid + off]; }
        __syncthreads();
    }
    if (tid == 0) {
        out[b * 2 + 0] = red0[0] + (float)T_ * b2[0];
        out[b * 2 + 1] = red1[0] + (float)T_ * b2[1];
    }
}

extern "C" void kernel_launch(void* const* d_in, const int* in_sizes, int n_in,
                              void* d_out, int out_size, void* d_ws, size_t ws_size,
                              hipStream_t stream) {
    (void)in_sizes; (void)n_in; (void)out_size; (void)ws_size;
    const float* x  = (const float*)d_in[0];   // [B,T,I]
    const float* w1 = (const float*)d_in[1];   // [H,I]
    const float* b1 = (const float*)d_in[2];   // [H]
    const float* w2 = (const float*)d_in[3];   // [O,H]
    const float* b2 = (const float*)d_in[4];   // [O]
    float* out = (float*)d_out;                // [B,O]

    char* ws = (char*)d_ws;
    __bf16* Xb = (__bf16*)ws;                                  // 67,108,864 B
    __bf16* Wb = (__bf16*)(ws + (size_t)B_ * T_ * I_ * 2);     //  4,194,304 B
    float*  S  = (float*) (ws + (size_t)B_ * T_ * I_ * 2
                              + (size_t)H_ * I_ * 2);          //  2,097,152 B

    k_convert_x<<<4096, 256, 0, stream>>>(x, Xb);
    k_convert_w<<<2048, 256, 0, stream>>>(w1, Wb);

    dim3 grid(B_ / 32, H_ / 64);   // 8 x 32 = 256 workgroups
    k_snn<<<grid, 256, 0, stream>>>(Xb, Wb, b1, S);

    k_out<<<B_, 256, 0, stream>>>(S, w2, b2, out);
}
